// Conv1DBlockSqueezeformer_23381801959738
// MI455X (gfx1250) — compile-verified, matrix operand order checked by reading, not run on hardware
//
#include <hip/hip_runtime.h>

// ---------------- problem constants ----------------
constexpr int Bb = 32;
constexpr int Tt = 2048;
constexpr int Cc = 384;          // C
constexpr int Ee = 1536;         // 4C
constexpr int Hh = 768;          // 2C (post-GLU channels)
constexpr int NR = Bb * Tt;      // 65536 rows
constexpr int DWK = 17;

// ---------------- bf16 helpers ----------------
typedef __attribute__((ext_vector_type(16))) __bf16 v16bf;
typedef __attribute__((ext_vector_type(8)))  float  v8f;

union Frag { v16bf v; uint4 q[2]; };

__device__ __forceinline__ unsigned short f2bf(float f) {
    union { float f; unsigned u; } c; c.f = f;
    unsigned u = c.u;
    unsigned r = (u + 0x7FFFu + ((u >> 16) & 1u)) >> 16;
    return (unsigned short)r;
}
__device__ __forceinline__ float bf2f(unsigned short h) {
    union { unsigned u; float f; } c; c.u = ((unsigned)h) << 16;
    return c.f;
}
__device__ __forceinline__ float silu(float x) { return x / (1.0f + __expf(-x)); }
__device__ __forceinline__ float sigm(float x) { return 1.0f / (1.0f + __expf(-x)); }

// ---------------- gfx1250 async global->LDS staging (guarded) ----------------
#if defined(__has_builtin)
#if __has_builtin(__builtin_amdgcn_global_load_async_to_lds_b128)
#define HAVE_ASYNC_LDS 1
#endif
#endif

#ifdef HAVE_ASYNC_LDS
typedef int v4i_vec __attribute__((vector_size(16)));
typedef v4i_vec __attribute__((address_space(1)))* g_v4i_p;
typedef v4i_vec __attribute__((address_space(3)))* l_v4i_p;

__device__ __forceinline__ void async_copy16(const void* g, void* l) {
    // per-lane 16B DMA: LDS[l] = GLOBAL[g], tracked by ASYNCcnt, no VGPR data
    __builtin_amdgcn_global_load_async_to_lds_b128((g_v4i_p)g, (l_v4i_p)l, 0, 0);
}
__device__ __forceinline__ void async_wait_all() {
#if __has_builtin(__builtin_amdgcn_s_wait_asynccnt)
    __builtin_amdgcn_s_wait_asynccnt(0);
#else
    asm volatile("s_wait_asynccnt 0x0" ::: "memory");
#endif
}
#endif

// A fragment from LDS panel (row-major bf16, stride in elements, 16B aligned)
__device__ __forceinline__ Frag frag_lds(const unsigned short* lds, int rowBase,
                                         int stride, int k0, int lane) {
    Frag f;
    int row = rowBase + (lane & 15);
    int kb  = k0 + ((lane >> 4) << 3);
    f.q[0] = *(const uint4*)(lds + row * stride + kb);
    f.q[1] = *(const uint4*)(lds + row * stride + kb + 16);
    return f;
}
// B fragment from a precomputed per-(wave,ntile) row base pointer:
// base = Wt + (n0 + lane%16)*K + (lane/16)*8 ; per k-step just add k0.
__device__ __forceinline__ Frag frag_b(const unsigned short* base, int k0) {
    Frag f;
    f.q[0] = *(const uint4*)(base + k0);
    f.q[1] = *(const uint4*)(base + k0 + 16);
    return f;
}
__device__ __forceinline__ const unsigned short* b_base(const unsigned short* wt,
                                                        int n0, int K, int lane) {
    return wt + (size_t)(n0 + (lane & 15)) * K + ((lane >> 4) << 3);
}
__device__ __forceinline__ v8f wmma_bf16(const Frag& a, const Frag& b, v8f c) {
    return __builtin_amdgcn_wmma_f32_16x16x32_bf16(false, a.v, false, b.v,
                                                   (short)0, c, false, false);
}

// ---------------- prologue kernels ----------------
__global__ void k_cvt(const float* __restrict__ in, unsigned short* __restrict__ out, int n) {
    int i = blockIdx.x * 256 + threadIdx.x;
    if (i < n) out[i] = f2bf(in[i]);
}
// in: f32 [K][N] row-major  ->  out: bf16 [N][K]
__global__ void k_tr_cvt(const float* __restrict__ in, unsigned short* __restrict__ out,
                         int K, int N) {
    int i = blockIdx.x * 256 + threadIdx.x;
    if (i >= K * N) return;
    int n = i / K, k = i % K;
    out[(size_t)n * K + k] = f2bf(in[(size_t)k * N + n]);
}
__global__ void k_zero(float* p, int n) {
    int i = blockIdx.x * 256 + threadIdx.x;
    if (i < n) p[i] = 0.0f;
}

// ---------------- GEMM + GLU (expand and ffn1; K = 384) ----------------
// out[m][n] = glu(A@W + bias), W^T rows [0..768) = 'a' half, [768..1536) = 'g' half.
// Block: 64 rows x 128 cols, 8 waves (4 M x 2 N), 4 n-tiles * 2 halves per wave.
__global__ __launch_bounds__(256)
void k_gemm_glu(const unsigned short* __restrict__ Abf,   // [NR][384] bf16
                const unsigned short* __restrict__ Wt,    // [1536][384] bf16
                const float* __restrict__ bias,           // [1536]
                unsigned short* __restrict__ Out) {       // [NR][768] bf16
    constexpr int K = 384, STR = K + 8;
    __shared__ __align__(16) unsigned short As[64 * STR];

    const int lane = threadIdx.x & 31;
    const int wid  = threadIdx.x >> 5;
    const int wm = wid >> 1, wn = wid & 1;
    const int m0 = blockIdx.x * 64;
    const int n0 = blockIdx.y * 128;

    // stage A panel (64 x 384 bf16 = 48KB)
    const uint4* src = (const uint4*)(Abf + (size_t)m0 * K);
#ifdef HAVE_ASYNC_LDS
    for (int i = threadIdx.x; i < 64 * (K / 8); i += 256) {
        int r = i / (K / 8), c = i % (K / 8);
        async_copy16(src + (size_t)r * (K / 8) + c, &As[r * STR + c * 8]);
    }
    async_wait_all();
#else
    for (int i = threadIdx.x; i < 64 * (K / 8); i += 256) {
        int r = i / (K / 8), c = i % (K / 8);
        *(uint4*)&As[r * STR + c * 8] = src[(size_t)r * (K / 8) + c];
    }
#endif
    __syncthreads();

    // hoisted per-n-tile B row bases (a-half and g-half of the GLU)
    const unsigned short* ba_p[4];
    const unsigned short* bg_p[4];
#pragma unroll
    for (int j = 0; j < 4; ++j) {
        int n = n0 + wn * 64 + j * 16;
        ba_p[j] = b_base(Wt, n,      K, lane);
        bg_p[j] = b_base(Wt, n + Hh, K, lane);
    }

    v8f acc_a[4] = {}, acc_g[4] = {};
    for (int kk = 0; kk < K; kk += 32) {
        Frag a = frag_lds(As, wm * 16, STR, kk, lane);
#pragma unroll
        for (int j = 0; j < 4; ++j) {
            Frag ba = frag_b(ba_p[j], kk);
            Frag bg = frag_b(bg_p[j], kk);
            acc_a[j] = wmma_bf16(a, ba, acc_a[j]);
            acc_g[j] = wmma_bf16(a, bg, acc_g[j]);
        }
    }

    const int rbase = m0 + wm * 16 + ((lane >> 4) << 3);
#pragma unroll
    for (int j = 0; j < 4; ++j) {
        int n = n0 + wn * 64 + j * 16 + (lane & 15);
        float b_a = bias[n], b_g = bias[n + Hh];
#pragma unroll
        for (int v = 0; v < 8; ++v) {
            float av = acc_a[j][v] + b_a;
            float gv = acc_g[j][v] + b_g;
            Out[(size_t)(rbase + v) * Hh + n] = f2bf(av * silu(gv));
        }
    }
}

// ---------------- depthwise conv + BN + SiLU + ECA partial sums ----------------
// grid (B, H/64, T/512); 256 threads = 64 channels x 4 time-segments of 128.
__global__ __launch_bounds__(256)
void k_dwconv(const unsigned short* __restrict__ H1, const float* __restrict__ dw_w,
              const float* __restrict__ bn_g, const float* __restrict__ bn_b,
              const float* __restrict__ bn_m, const float* __restrict__ bn_v,
              unsigned short* __restrict__ H2, float* __restrict__ Sums) {
    const int c    = blockIdx.y * 64 + (threadIdx.x & 63);
    const int b    = blockIdx.x;
    const int t0   = blockIdx.z * 512 + (threadIdx.x >> 6) * 128;

    float w[DWK];
#pragma unroll
    for (int j = 0; j < DWK; ++j) w[j] = dw_w[c * DWK + j];
    float inv = rsqrtf(bn_v[c] + 1e-5f);
    float sc  = bn_g[c] * inv;
    float sh  = bn_b[c] - bn_m[c] * sc;

    const size_t base = ((size_t)b * Tt) * Hh + c;
    float win[DWK];
#pragma unroll
    for (int j = 0; j < DWK; ++j) {
        int t = t0 - 8 + j;
        win[j] = (t >= 0 && t < Tt) ? bf2f(H1[base + (size_t)t * Hh]) : 0.0f;
    }
    float lsum = 0.0f;
    for (int t = t0; t < t0 + 128; ++t) {
        float a = 0.0f;
#pragma unroll
        for (int j = 0; j < DWK; ++j) a += w[j] * win[j];
        a = a * sc + sh;
        float v = silu(a);
        H2[base + (size_t)t * Hh] = f2bf(v);
        lsum += v;
#pragma unroll
        for (int j = 0; j < DWK - 1; ++j) win[j] = win[j + 1];
        int tn = t + 9;
        win[DWK - 1] = (tn < Tt) ? bf2f(H1[base + (size_t)tn * Hh]) : 0.0f;
    }
    atomicAdd(&Sums[(size_t)b * Hh + c], lsum);
}

// ---------------- ECA gate ----------------
__global__ void k_eca(const float* __restrict__ Sums, const float* __restrict__ ew,
                      float* __restrict__ Gate) {
    int i = blockIdx.x * 256 + threadIdx.x;
    if (i >= Bb * Hh) return;
    int b = i / Hh, c = i % Hh;
    float s = 0.0f;
#pragma unroll
    for (int j = 0; j < 5; ++j) {
        int cc = c + j - 2;
        if (cc >= 0 && cc < Hh) s += ew[j] * (Sums[b * Hh + cc] * (1.0f / Tt));
    }
    Gate[i] = sigm(s);
}

// ---------------- proj GEMM (K=768, N=384): gate folded into A staging ----------------
// Block: 32 rows x 128 cols, 8 waves (2 M x 4 N), 2 n-tiles per wave.
// Staging stays synchronous: the ECA gate multiply is fused into the LDS transform.
__global__ __launch_bounds__(256)
void k_gemm_proj(const unsigned short* __restrict__ H2,   // [NR][768] bf16
                 const unsigned short* __restrict__ Wt,   // [384][768] bf16
                 const float* __restrict__ Gate,          // [B][768]
                 const float* __restrict__ proj_b, const float* __restrict__ sb1_s,
                 const float* __restrict__ sb1_b, const float* __restrict__ Xf,
                 float* __restrict__ R32, unsigned short* __restrict__ Rbf) {
    constexpr int K = 768, STR = K + 8;
    __shared__ __align__(16) unsigned short As[32 * STR];

    const int lane = threadIdx.x & 31;
    const int wid  = threadIdx.x >> 5;
    const int wm = wid >> 2, wn = wid & 3;
    const int m0 = blockIdx.x * 32;
    const int n0 = blockIdx.y * 128;

    const float* gb = Gate + (size_t)(m0 / Tt) * Hh;  // whole tile in one batch
    for (int i = threadIdx.x; i < 32 * (K / 8); i += 256) {
        int r = i / (K / 8), c = i % (K / 8);
        uint4 raw = *(const uint4*)(H2 + (size_t)(m0 + r) * K + c * 8);
        const unsigned short* p = (const unsigned short*)&raw;
        unsigned short o[8];
#pragma unroll
        for (int e = 0; e < 8; ++e) o[e] = f2bf(bf2f(p[e]) * gb[c * 8 + e]);
        *(uint4*)&As[r * STR + c * 8] = *(const uint4*)o;
    }
    __syncthreads();

    const unsigned short* b_p[2];
#pragma unroll
    for (int j = 0; j < 2; ++j)
        b_p[j] = b_base(Wt, n0 + wn * 32 + j * 16, K, lane);

    v8f acc[2] = {};
    for (int kk = 0; kk < K; kk += 32) {
        Frag a = frag_lds(As, wm * 16, STR, kk, lane);
#pragma unroll
        for (int j = 0; j < 2; ++j) {
            Frag bfr = frag_b(b_p[j], kk);
            acc[j] = wmma_bf16(a, bfr, acc[j]);
        }
    }

    const int rbase = m0 + wm * 16 + ((lane >> 4) << 3);
#pragma unroll
    for (int j = 0; j < 2; ++j) {
        int n = n0 + wn * 32 + j * 16 + (lane & 15);
        float pb = proj_b[n], sc = sb1_s[n], sb = sb1_b[n];
#pragma unroll
        for (int v = 0; v < 8; ++v) {
            size_t idx = (size_t)(rbase + v) * Cc + n;
            float r = (acc[j][v] + pb) * sc + sb + Xf[idx];
            R32[idx] = r;
            Rbf[idx] = f2bf(r);
        }
    }
}

// ---------------- ffn2 GEMM (K=768) + sb2 + residual + LayerNorm ----------------
// Block covers 16 rows x all 384 cols so LN stays in-block. 8 waves, 3 n-tiles each.
__global__ __launch_bounds__(256)
void k_gemm_ffn2_ln(const unsigned short* __restrict__ F1,  // [NR][768] bf16
                    const unsigned short* __restrict__ Wt,  // [384][768] bf16
                    const float* __restrict__ b2, const float* __restrict__ sb2_s,
                    const float* __restrict__ sb2_b, const float* __restrict__ R32,
                    const float* __restrict__ ln_g, const float* __restrict__ ln_b,
                    float* __restrict__ Out) {
    constexpr int K = 768, STR = K + 8;
    __shared__ __align__(16) unsigned short As[16 * STR];
    __shared__ __align__(16) float Zs[16 * Cc];

    const int lane = threadIdx.x & 31;
    const int wid  = threadIdx.x >> 5;
    const int m0 = blockIdx.x * 16;

    const uint4* src = (const uint4*)(F1 + (size_t)m0 * K);
#ifdef HAVE_ASYNC_LDS
    for (int i = threadIdx.x; i < 16 * (K / 8); i += 256) {
        int r = i / (K / 8), c = i % (K / 8);
        async_copy16(src + (size_t)r * (K / 8) + c, &As[r * STR + c * 8]);
    }
    async_wait_all();
#else
    for (int i = threadIdx.x; i < 16 * (K / 8); i += 256) {
        int r = i / (K / 8), c = i % (K / 8);
        *(uint4*)&As[r * STR + c * 8] = src[(size_t)r * (K / 8) + c];
    }
#endif
    __syncthreads();

    const unsigned short* b_p[3];
#pragma unroll
    for (int j = 0; j < 3; ++j)
        b_p[j] = b_base(Wt, wid * 48 + j * 16, K, lane);

    v8f acc[3] = {};
    for (int kk = 0; kk < K; kk += 32) {
        Frag a = frag_lds(As, 0, STR, kk, lane);
#pragma unroll
        for (int j = 0; j < 3; ++j) {
            Frag bfr = frag_b(b_p[j], kk);
            acc[j] = wmma_bf16(a, bfr, acc[j]);
        }
    }

    const int mloc = (lane >> 4) << 3;
#pragma unroll
    for (int j = 0; j < 3; ++j) {
        int n = wid * 48 + j * 16 + (lane & 15);
        float bb = b2[n], sc = sb2_s[n], sb = sb2_b[n];
#pragma unroll
        for (int v = 0; v < 8; ++v) {
            int ml = mloc + v;
            float z = (acc[j][v] + bb) * sc + sb + R32[(size_t)(m0 + ml) * Cc + n];
            Zs[ml * Cc + n] = z;
        }
    }
    __syncthreads();

    for (int rr = wid; rr < 16; rr += 8) {
        const float* zr = &Zs[rr * Cc];
        float s = 0.0f;
        for (int i = lane; i < Cc; i += 32) s += zr[i];
#pragma unroll
        for (int off = 16; off; off >>= 1) s += __shfl_xor(s, off, 32);
        float mu = s * (1.0f / Cc);
        float vs = 0.0f;
        for (int i = lane; i < Cc; i += 32) { float d = zr[i] - mu; vs += d * d; }
#pragma unroll
        for (int off = 16; off; off >>= 1) vs += __shfl_xor(vs, off, 32);
        float inv = rsqrtf(vs * (1.0f / Cc) + 1e-6f);
        float* orow = Out + (size_t)(m0 + rr) * Cc;
        for (int i = lane; i < Cc; i += 32)
            orow[i] = (zr[i] - mu) * inv * ln_g[i] + ln_b[i];
    }
}

// ---------------- host launch ----------------
extern "C" void kernel_launch(void* const* d_in, const int* in_sizes, int n_in,
                              void* d_out, int out_size, void* d_ws, size_t ws_size,
                              hipStream_t stream) {
    const float* x        = (const float*)d_in[0];
    const float* expand_w = (const float*)d_in[1];
    const float* expand_b = (const float*)d_in[2];
    const float* dw_w     = (const float*)d_in[3];
    const float* bn_g     = (const float*)d_in[4];
    const float* bn_b     = (const float*)d_in[5];
    const float* bn_m     = (const float*)d_in[6];
    const float* bn_v     = (const float*)d_in[7];
    const float* eca_w    = (const float*)d_in[8];
    const float* proj_w   = (const float*)d_in[9];
    const float* proj_b   = (const float*)d_in[10];
    const float* sb1_s    = (const float*)d_in[11];
    const float* sb1_b    = (const float*)d_in[12];
    const float* ffn_w1   = (const float*)d_in[13];
    const float* ffn_b1   = (const float*)d_in[14];
    const float* ffn_w2   = (const float*)d_in[15];
    const float* ffn_b2   = (const float*)d_in[16];
    const float* sb2_s    = (const float*)d_in[17];
    const float* sb2_b    = (const float*)d_in[18];
    const float* ln_g     = (const float*)d_in[19];
    const float* ln_b     = (const float*)d_in[20];
    float* out = (float*)d_out;

    char* ws = (char*)d_ws;
    size_t off = 0;
    auto take = [&](size_t bytes) -> void* {
        void* p = (void*)(ws + off);
        off += (bytes + 255) & ~(size_t)255;
        return p;
    };
    unsigned short* xbf  = (unsigned short*)take((size_t)NR * Cc * 2);
    unsigned short* h1   = (unsigned short*)take((size_t)NR * Hh * 2);
    unsigned short* h2   = (unsigned short*)take((size_t)NR * Hh * 2);
    float*          r32  = (float*)take((size_t)NR * Cc * 4);
    unsigned short* wtE  = (unsigned short*)take((size_t)Ee * Cc * 2);
    unsigned short* wtP  = (unsigned short*)take((size_t)Cc * Hh * 2);
    unsigned short* wtF1 = (unsigned short*)take((size_t)Ee * Cc * 2);
    unsigned short* wtF2 = (unsigned short*)take((size_t)Cc * Hh * 2);
    float*          sums = (float*)take((size_t)Bb * Hh * 4);
    float*          gate = (float*)take((size_t)Bb * Hh * 4);
    unsigned short* rbf  = xbf;  // xbf dead after expand GEMM
    unsigned short* f1   = h1;   // h1 dead after dwconv

    // prologue: precision conversion + weight transposes (tiny, L2-resident)
    int n = NR * Cc;
    k_cvt<<<(n + 255) / 256, 256, 0, stream>>>(x, xbf, n);
    k_tr_cvt<<<(Cc * Ee + 255) / 256, 256, 0, stream>>>(expand_w, wtE, Cc, Ee);
    k_tr_cvt<<<(Hh * Cc + 255) / 256, 256, 0, stream>>>(proj_w,  wtP, Hh, Cc);
    k_tr_cvt<<<(Cc * Ee + 255) / 256, 256, 0, stream>>>(ffn_w1,  wtF1, Cc, Ee);
    k_tr_cvt<<<(Hh * Cc + 255) / 256, 256, 0, stream>>>(ffn_w2,  wtF2, Hh, Cc);
    k_zero<<<(Bb * Hh + 255) / 256, 256, 0, stream>>>(sums, Bb * Hh);

    // expand + GLU
    k_gemm_glu<<<dim3(NR / 64, Hh / 128), 256, 0, stream>>>(xbf, wtE, expand_b, h1);
    // depthwise conv + BN + SiLU + ECA partial sums
    k_dwconv<<<dim3(Bb, Hh / 64, Tt / 512), 256, 0, stream>>>(h1, dw_w, bn_g, bn_b,
                                                              bn_m, bn_v, h2, sums);
    // ECA gate
    k_eca<<<(Bb * Hh + 255) / 256, 256, 0, stream>>>(sums, eca_w, gate);
    // proj (gated A) + sb1 + residual
    k_gemm_proj<<<dim3(NR / 32, Cc / 128), 256, 0, stream>>>(h2, wtP, gate, proj_b,
                                                             sb1_s, sb1_b, x, r32, rbf);
    // ffn1 + GLU
    k_gemm_glu<<<dim3(NR / 64, Hh / 128), 256, 0, stream>>>(rbf, wtF1, ffn_b1, f1);
    // ffn2 + sb2 + residual + LayerNorm
    k_gemm_ffn2_ln<<<NR / 16, 256, 0, stream>>>(f1, wtF2, ffn_b2, sb2_s, sb2_b,
                                                r32, ln_g, ln_b, out);
}
